// SelfAggregationModuleV2_23776938951341
// MI455X (gfx1250) — matrix operand order repaired and checked by reading, not compiled, not run on hardware
//
#include <hip/hip_runtime.h>

// ---------------------------------------------------------------------------
// Types for WMMA fragments (wave32, CDNA5)
// ---------------------------------------------------------------------------
typedef __attribute__((ext_vector_type(16))) __bf16        v16bf;
typedef __attribute__((ext_vector_type(8)))  float         v8f;
typedef __attribute__((ext_vector_type(2)))  float         v2f;
typedef __attribute__((ext_vector_type(4)))  unsigned int  v4u;
typedef unsigned short u16;

union FragBF {
    v16bf bf;          // operand for v_wmma_f32_16x16x32_bf16
    u16   u[16];
    unsigned int w[8]; // bf16 pairs, for packed-math unpacking
    v4u   q[2];        // two 16-byte chunks for vector loads/stores
};

static __device__ __forceinline__ u16 f2bf(float f) {
    unsigned int u = __float_as_uint(f);
    u += 0x7FFFu + ((u >> 16) & 1u);   // round-to-nearest-even
    return (u16)(u >> 16);
}
static __device__ __forceinline__ float bf2f(u16 h) {
    return __uint_as_float(((unsigned int)h) << 16);
}
// Unpack a bf16 pair (one dword) into a packed float2 in two VALU ops.
static __device__ __forceinline__ v2f bfpair2f(unsigned int w) {
    v2f r;
    r.x = __uint_as_float(w << 16);
    r.y = __uint_as_float(w & 0xFFFF0000u);
    return r;
}

// Problem constants
#define C_DIM   128
#define H_DIM   64
#define W_DIM   64
#define Z_DIM   8
#define NSPAT   (H_DIM * W_DIM * Z_DIM)   // 32768
#define NHEADS  8
#define HD      16
#define QKV3    (3 * C_DIM)               // 384
#define KTILES  (C_DIM / 32)              // 4

// ---------------------------------------------------------------------------
// 1) Per-channel mean / rsqrt(var+eps) over spatial dims
// ---------------------------------------------------------------------------
__global__ __launch_bounds__(256)
void nat3d_stats_kernel(const float* __restrict__ x,
                        float* __restrict__ mu, float* __restrict__ rs) {
    const int c = blockIdx.x;
    const float* p = x + (size_t)c * NSPAT;
    float s = 0.f, s2 = 0.f;
    for (int i = threadIdx.x; i < NSPAT; i += 256) {
        float v = p[i];
        s += v; s2 += v * v;
    }
    __shared__ float sh[256], sh2[256];
    sh[threadIdx.x] = s; sh2[threadIdx.x] = s2;
    __syncthreads();
    for (int off = 128; off > 0; off >>= 1) {
        if (threadIdx.x < off) {
            sh[threadIdx.x]  += sh[threadIdx.x + off];
            sh2[threadIdx.x] += sh2[threadIdx.x + off];
        }
        __syncthreads();
    }
    if (threadIdx.x == 0) {
        float m   = sh[0] * (1.0f / NSPAT);
        float var = sh2[0] * (1.0f / NSPAT) - m * m;
        mu[c] = m;
        rs[c] = rsqrtf(var + 1e-5f);
    }
}

// ---------------------------------------------------------------------------
// 2) Pack weights directly into WMMA B-fragment layout (bf16).
//    Packed index: ((nt*KTILES + kt)*32 + lane)*16 + e
//    maps to W[k * pitch + col] with
//      col   = nt*16 + (lane & 15)
//      khalf = (lane < 16) ? 0 : 8
//      k     = kt*32 + khalf + (e < 8 ? e : 8 + (e - 8) + 8)
//    GEMM side then loads each fragment as two coalesced b128 loads.
// ---------------------------------------------------------------------------
static __device__ __forceinline__ u16 pack_one(const float* __restrict__ W,
                                               int pitch, int idx) {
    const int e    = idx & 15;
    const int lane = (idx >> 4) & 31;
    const int kt   = (idx >> 9) & (KTILES - 1);
    const int nt   = idx >> (9 + 2);            // KTILES == 4
    const int col  = nt * 16 + (lane & 15);
    const int khalf = (lane < 16) ? 0 : 8;
    const int k    = kt * 32 + khalf + ((e < 8) ? e : (e + 8));
    return f2bf(W[(size_t)k * pitch + col]);
}

__global__ __launch_bounds__(256)
void nat3d_wpack_kernel(const float* __restrict__ wqkv,
                        const float* __restrict__ wproj,
                        u16* __restrict__ wqkv_pk, u16* __restrict__ wproj_pk) {
    int idx = blockIdx.x * 256 + threadIdx.x;   // 49152 + 16384 = 65536 total
    if (idx < C_DIM * QKV3) {
        wqkv_pk[idx] = pack_one(wqkv, QKV3, idx);
    } else {
        int j = idx - C_DIM * QKV3;
        wproj_pk[j] = pack_one(wproj, C_DIM, j);
    }
}

// ---------------------------------------------------------------------------
// 3) Normalize + transpose (C-major -> spatial-major) into bf16 [NSPAT][128]
// ---------------------------------------------------------------------------
__global__ __launch_bounds__(256)
void nat3d_transnorm_kernel(const float* __restrict__ x,
                            const float* __restrict__ mu,
                            const float* __restrict__ rs,
                            u16* __restrict__ xn) {
    __shared__ u16 tile[32][33];
    const int n0 = blockIdx.x * 32;    // spatial tile
    const int c0 = blockIdx.y * 32;    // channel tile
    {
        const int nn = threadIdx.x & 31;
        const int cb = threadIdx.x >> 5;  // 0..7
        #pragma unroll
        for (int i = 0; i < 4; ++i) {
            int cc = cb + 8 * i;
            int c  = c0 + cc;
            float v = (x[(size_t)c * NSPAT + (n0 + nn)] - mu[c]) * rs[c];
            tile[cc][nn] = f2bf(v);
        }
    }
    __syncthreads();
    {
        const int cc = threadIdx.x & 31;
        const int nb = threadIdx.x >> 5;
        #pragma unroll
        for (int i = 0; i < 4; ++i) {
            int nn = nb + 8 * i;
            xn[(size_t)(n0 + nn) * C_DIM + (c0 + cc)] = tile[cc][nn];
        }
    }
}

// ---------------------------------------------------------------------------
// Fragment loaders — both A and packed-B are two b128 loads per lane.
// ---------------------------------------------------------------------------
static __device__ __forceinline__ FragBF load_a_frag(const u16* __restrict__ A,
                                                     int row, int k0, int lane) {
    FragBF a;
    const int khalf = (lane < 16) ? 0 : 8;
    const u16* ap = A + (size_t)row * C_DIM + k0 + khalf;
    a.q[0] = *(const v4u*)(ap);        // K offsets 0..7  of this half
    a.q[1] = *(const v4u*)(ap + 16);   // K offsets 16..23 of this half
    return a;
}

static __device__ __forceinline__ FragBF load_b_frag_pk(const u16* __restrict__ Wpk,
                                                        int nt, int kt, int lane) {
    FragBF b;
    const u16* bp = Wpk + ((size_t)(nt * KTILES + kt) * 32 + lane) * 16;
    b.q[0] = *(const v4u*)(bp);
    b.q[1] = *(const v4u*)(bp + 8);
    return b;
}

// ---------------------------------------------------------------------------
// 4) QKV GEMM:  qkv[32768][384] = xn[32768][128] @ w_qkv[128][384] + b_qkv
//    Block = 32 rows, 8 waves; wave covers 48 cols -> 2x3 accumulators,
//    24 WMMAs per wave.
// ---------------------------------------------------------------------------
__global__ __launch_bounds__(256)
void nat3d_qkv_gemm_kernel(const u16* __restrict__ A,
                           const u16* __restrict__ Wpk,
                           const float* __restrict__ bias,
                           u16* __restrict__ Out) {
    const int wave = threadIdx.x >> 5;
    const int lane = threadIdx.x & 31;
    const int m0 = blockIdx.x * 32;
    const int nt0 = wave * 3;          // first of 3 n-tiles (48 cols)
    const int lrow = lane & 15;

    v8f acc[2][3] = {};
    #pragma unroll
    for (int kt = 0; kt < KTILES; ++kt) {
        const int k0 = kt * 32;
        FragBF a0 = load_a_frag(A, m0 + lrow,      k0, lane);
        FragBF a1 = load_a_frag(A, m0 + 16 + lrow, k0, lane);
        #pragma unroll
        for (int t = 0; t < 3; ++t) {
            FragBF b = load_b_frag_pk(Wpk, nt0 + t, kt, lane);
            acc[0][t] = __builtin_amdgcn_wmma_f32_16x16x32_bf16(
                false, a0.bf, false, b.bf, (short)0, acc[0][t], false, false);
            acc[1][t] = __builtin_amdgcn_wmma_f32_16x16x32_bf16(
                false, a1.bf, false, b.bf, (short)0, acc[1][t], false, false);
        }
    }
    const int mh = (lane >= 16) ? 8 : 0;
    #pragma unroll
    for (int t = 0; t < 3; ++t) {
        const int n = (nt0 + t) * 16 + lrow;
        const float bi = bias[n];
        #pragma unroll
        for (int mt = 0; mt < 2; ++mt) {
            #pragma unroll
            for (int r = 0; r < 8; ++r) {
                int m = m0 + mt * 16 + r + mh;
                Out[(size_t)m * QKV3 + n] = f2bf(acc[mt][t][r] + bi);
            }
        }
    }
}

// ---------------------------------------------------------------------------
// 5) Neighborhood attention: one thread per (voxel, head).
//    Dot products and value accumulation use packed float2 math so the
//    MAC chains lower to v_pk_fma_f32 (2 FLOPs/VALU instruction).
// ---------------------------------------------------------------------------
__global__ __launch_bounds__(256)
void nat3d_attn_kernel(const u16* __restrict__ qkv,
                       const float* __restrict__ rpb,
                       u16* __restrict__ out) {
    const int t = blockIdx.x * 256 + threadIdx.x;
    const int h = t & 7;
    const int n = t >> 3;
    const int z = n & 7;
    const int w = (n >> 3) & 63;
    const int y = n >> 9;

    v2f q2[8];
    {
        const u16* qp = qkv + (size_t)n * QKV3 + h * HD;
        FragBF qf; qf.q[0] = *(const v4u*)qp; qf.q[1] = *(const v4u*)(qp + 8);
        #pragma unroll
        for (int j = 0; j < 8; ++j) {
            v2f p = bfpair2f(qf.w[j]);
            q2[j].x = p.x * 0.25f;   // hd^-0.5
            q2[j].y = p.y * 0.25f;
        }
    }

    const int sy = min(max(y - 1, 0), H_DIM - 3);
    const int sw = min(max(w - 1, 0), W_DIM - 3);
    const int sz = min(max(z - 1, 0), Z_DIM - 3);

    float logits[27];
    float mx = -1e30f;
    #pragma unroll
    for (int a = 0; a < 3; ++a) {
        #pragma unroll
        for (int b = 0; b < 3; ++b) {
            #pragma unroll
            for (int c = 0; c < 3; ++c) {
                const int yy = sy + a, ww = sw + b, zz = sz + c;
                const int nb = (yy * W_DIM + ww) * Z_DIM + zz;
                const u16* kp = qkv + (size_t)nb * QKV3 + C_DIM + h * HD;
                FragBF kf; kf.q[0] = *(const v4u*)kp; kf.q[1] = *(const v4u*)(kp + 8);
                v2f d2 = {0.f, 0.f};
                #pragma unroll
                for (int j = 0; j < 8; ++j) {
                    d2 = __builtin_elementwise_fma(q2[j], bfpair2f(kf.w[j]), d2);
                }
                const int rh = yy - y + 2, rw = ww - w + 2, rz = zz - z + 2;
                float l = d2.x + d2.y + rpb[((h * 5 + rh) * 5 + rw) * 5 + rz];
                logits[(a * 3 + b) * 3 + c] = l;
                mx = fmaxf(mx, l);
            }
        }
    }
    float se = 0.f;
    #pragma unroll
    for (int j = 0; j < 27; ++j) {
        float e = __expf(logits[j] - mx);
        logits[j] = e; se += e;
    }
    const float inv = 1.0f / se;

    v2f o2[8];
    #pragma unroll
    for (int j = 0; j < 8; ++j) o2[j] = (v2f){0.f, 0.f};
    #pragma unroll
    for (int a = 0; a < 3; ++a) {
        #pragma unroll
        for (int b = 0; b < 3; ++b) {
            #pragma unroll
            for (int c = 0; c < 3; ++c) {
                const int yy = sy + a, ww = sw + b, zz = sz + c;
                const int nb = (yy * W_DIM + ww) * Z_DIM + zz;
                const u16* vp = qkv + (size_t)nb * QKV3 + 2 * C_DIM + h * HD;
                FragBF vf; vf.q[0] = *(const v4u*)vp; vf.q[1] = *(const v4u*)(vp + 8);
                const float wgt = logits[(a * 3 + b) * 3 + c];
                const v2f wg = {wgt, wgt};
                #pragma unroll
                for (int j = 0; j < 8; ++j) {
                    o2[j] = __builtin_elementwise_fma(wg, bfpair2f(vf.w[j]), o2[j]);
                }
            }
        }
    }
    FragBF of;
    #pragma unroll
    for (int j = 0; j < 8; ++j) {
        of.u[2 * j]     = f2bf(o2[j].x * inv);
        of.u[2 * j + 1] = f2bf(o2[j].y * inv);
    }
    u16* op = out + (size_t)n * C_DIM + h * HD;
    *(v4u*)op       = of.q[0];
    *(v4u*)(op + 8) = of.q[1];
}

// ---------------------------------------------------------------------------
// 6) Projection GEMM + transposed store:
//    out[c][n] = sum_k attn[n][k] * w_proj[k][c] + b_proj[c]
//    Block = 32 rows, 8 waves; wave covers 16 cols -> 2 accumulators,
//    8 WMMAs per wave. Stores land contiguous along the spatial dim.
// ---------------------------------------------------------------------------
__global__ __launch_bounds__(256)
void nat3d_proj_gemm_kernel(const u16* __restrict__ A,
                            const u16* __restrict__ Wpk,
                            const float* __restrict__ bias,
                            float* __restrict__ Out) {
    const int wave = threadIdx.x >> 5;
    const int lane = threadIdx.x & 31;
    const int m0 = blockIdx.x * 32;
    const int nt = wave;               // one n-tile per wave (8 x 16 = 128)
    const int lrow = lane & 15;

    v8f acc[2] = {};
    #pragma unroll
    for (int kt = 0; kt < KTILES; ++kt) {
        const int k0 = kt * 32;
        FragBF a0 = load_a_frag(A, m0 + lrow,      k0, lane);
        FragBF a1 = load_a_frag(A, m0 + 16 + lrow, k0, lane);
        FragBF b  = load_b_frag_pk(Wpk, nt, kt, lane);
        acc[0] = __builtin_amdgcn_wmma_f32_16x16x32_bf16(
            false, a0.bf, false, b.bf, (short)0, acc[0], false, false);
        acc[1] = __builtin_amdgcn_wmma_f32_16x16x32_bf16(
            false, a1.bf, false, b.bf, (short)0, acc[1], false, false);
    }
    const int mh = (lane >= 16) ? 8 : 0;
    const int n = nt * 16 + lrow;              // output channel
    const float bi = bias[n];
    #pragma unroll
    for (int mt = 0; mt < 2; ++mt) {
        #pragma unroll
        for (int r = 0; r < 8; ++r) {
            int m = m0 + mt * 16 + r + mh;     // spatial index
            Out[(size_t)n * NSPAT + m] = acc[mt][r] + bi;
        }
    }
}

// ---------------------------------------------------------------------------
// Host-side launch
// ---------------------------------------------------------------------------
extern "C" void kernel_launch(void* const* d_in, const int* in_sizes, int n_in,
                              void* d_out, int out_size, void* d_ws, size_t ws_size,
                              hipStream_t stream) {
    (void)in_sizes; (void)n_in; (void)out_size; (void)ws_size;
    const float* x      = (const float*)d_in[0];   // (1,128,64,64,8)
    const float* w_qkv  = (const float*)d_in[1];   // (128,384)
    const float* b_qkv  = (const float*)d_in[2];   // (384,)
    const float* rpb    = (const float*)d_in[3];   // (8,5,5,5)
    const float* w_proj = (const float*)d_in[4];   // (128,128)
    const float* b_proj = (const float*)d_in[5];   // (128,)
    float* out = (float*)d_out;

    char* ws = (char*)d_ws;
    float* mu      = (float*)(ws + 0);                      //   512 B
    float* rs      = (float*)(ws + 512);                    //   512 B
    u16*   wqkv_pk = (u16*)  (ws + 4096);                   //  96 KB packed
    u16*   wproj_pk= (u16*)  (ws + 4096 + 98304);           //  32 KB packed
    u16*   xn      = (u16*)  (ws + 262144);                 //   8 MB
    u16*   qkvbuf  = (u16*)  (ws + 262144 + 8388608);       //  24 MB
    u16*   attnbuf = (u16*)  (ws + 262144 + 8388608 + 25165824); // 8 MB
    // total workspace use: ~42.2 MB

    nat3d_stats_kernel<<<C_DIM, 256, 0, stream>>>(x, mu, rs);
    nat3d_wpack_kernel<<<(C_DIM * QKV3 + C_DIM * C_DIM) / 256, 256, 0, stream>>>(
        w_qkv, w_proj, wqkv_pk, wproj_pk);
    nat3d_transnorm_kernel<<<dim3(NSPAT / 32, C_DIM / 32), 256, 0, stream>>>(
        x, mu, rs, xn);
    nat3d_qkv_gemm_kernel<<<NSPAT / 32, 256, 0, stream>>>(
        xn, wqkv_pk, b_qkv, qkvbuf);
    nat3d_attn_kernel<<<(NSPAT * NHEADS) / 256, 256, 0, stream>>>(
        qkvbuf, rpb, attnbuf);
    nat3d_proj_gemm_kernel<<<NSPAT / 32, 256, 0, stream>>>(
        attnbuf, wproj_pk, b_proj, out);
}
